// ASR_20581483282783
// MI455X (gfx1250) — matrix operand order repaired and checked by reading, not compiled, not run on hardware
//
#include <hip/hip_runtime.h>
#include <hip/hip_bf16.h>

// ---------------------------------------------------------------------------
// ASR stack (Conv1D s2 + 2x BiLSTM(512) + Dense) for MI455X (gfx1250, wave32).
// All matmuls run on v_wmma_f32_16x16x32_bf16 (bf16 inputs, f32 accumulate).
// Workspace requirement: ~750 MB (xg buffers dominate: 2 x 262 MB, reused
// across the two BiLSTM layers).
//
// Round-1 fix: k_gemm inner loop is now straight-line (no per-subtile branch
// around the WMMAs). Previous version's divergent `continue` caused exec-mask
// branches inside the K-loop and full accumulator spills to scratch.
// ---------------------------------------------------------------------------

typedef __attribute__((ext_vector_type(16))) __bf16 v16bf;
typedef __attribute__((ext_vector_type(8)))  __bf16 v8bf;
typedef __attribute__((ext_vector_type(8)))  float  v8f;

#define B_    32
#define T_IN  2000
#define CIN_  80
#define F_    256
#define T2_   1000
#define H_    512
#define G4_   2048           // 4*H (gate width)
#define V_    29
#define KC_   880            // 11*80 im2col K
#define KCP_  896            // padded to multiple of 32
#define MROWS 32000          // B_*T2_

#define ZLD_  2056           // zbuf row stride (f32), padded vs 2048
#define HLD_  520            // hbuf row stride (bf16), padded vs 512

__device__ __forceinline__ float sigmoidf_(float x) {
    return 1.0f / (1.0f + __expf(-x));
}

// A-fragment (16x32 bf16, M per lane): per ISA 7.12.2 — lane<16 holds
// K = kb+{0..7, 16..23}, lane>=16 holds K = kb+{8..15, 24..31}.
// With row-major [M][K] source: two contiguous 16B loads at +koffA and +koffA+16.
__device__ __forceinline__ v16bf fragA_ld(const __bf16* p) {
    v8bf lo = *(const v8bf*)(p);
    v8bf hi = *(const v8bf*)(p + 16);
    v16bf r;
#pragma unroll
    for (int i = 0; i < 8; ++i) { r[i] = lo[i]; r[i + 8] = hi[i]; }
    return r;
}

// B-fragment (32x16 bf16, N per lane): same K order as A — lane<16 holds
// K = kb+{0..7, 16..23}, lane>=16 holds K = kb+{8..15, 24..31}.
// With W^T stored [N][K] row-major: two contiguous 16B loads at +koffB and +koffB+16.
__device__ __forceinline__ v16bf fragB_ld(const __bf16* p) {
    v8bf lo = *(const v8bf*)(p);
    v8bf hi = *(const v8bf*)(p + 16);
    v16bf r;
#pragma unroll
    for (int i = 0; i < 8; ++i) { r[i] = lo[i]; r[i + 8] = hi[i]; }
    return r;
}

// ---------------------------------------------------------------------------
// Weight convert+transpose: src f32 [K][N] -> dst bf16 [Npad][Kpad], zero pad.
// ---------------------------------------------------------------------------
__global__ void k_transpose_bf16(const float* __restrict__ src, __bf16* __restrict__ dst,
                                 int K, int N, int Kpad, int Npad) {
    int idx = blockIdx.x * blockDim.x + threadIdx.x;
    int total = Kpad * Npad;
    if (idx >= total) return;
    int n = idx / Kpad;
    int k = idx - n * Kpad;
    float v = (k < K && n < N) ? src[(size_t)k * N + n] : 0.0f;
    dst[(size_t)n * Kpad + k] = (__bf16)v;
}

// ---------------------------------------------------------------------------
// im2col for Conv1D (stride 2, SAME: pad_left=4) -> bf16 [32000][896]
// ---------------------------------------------------------------------------
__global__ void k_im2col(const float* __restrict__ x, __bf16* __restrict__ out) {
    int idx = blockIdx.x * blockDim.x + threadIdx.x;
    if (idx >= MROWS * KCP_) return;
    int row = idx / KCP_;
    int kk  = idx - row * KCP_;
    float v = 0.0f;
    if (kk < KC_) {
        int b  = row / T2_;
        int tq = row - b * T2_;
        int k  = kk / CIN_;
        int c  = kk - k * CIN_;
        int tin = 2 * tq + k - 4;
        if (tin >= 0 && tin < T_IN) v = x[((size_t)b * T_IN + tin) * CIN_ + c];
    }
    out[(size_t)row * KCP_ + kk] = (__bf16)v;
}

// ---------------------------------------------------------------------------
// Generic WMMA GEMM: C[M][N] = A[M][K] * BT[N][K]^T + bias, optional ReLU.
// block = 256 threads (8 waves, 4x2), tile 64(M) x 128(N).
// Out-of-range N-subtiles (dense layer only) are handled by CLAMPING the
// B-row pointer (loads stay in-bounds, math is discarded in the epilogue) so
// the K-loop carries no control flow at all.
// Writes f32 (outF, ld=N) and/or bf16 (outH, ld=Npad).
// ---------------------------------------------------------------------------
__global__ void k_gemm(const __bf16* __restrict__ A, const __bf16* __restrict__ BT,
                       const float* __restrict__ bias, float* __restrict__ outF,
                       __bf16* __restrict__ outH, int N, int Npad, int K, int relu) {
    const int wid  = threadIdx.x >> 5;
    const int lane = threadIdx.x & 31;
    const int wm   = wid & 3;
    const int wn   = wid >> 2;
    const int m0   = blockIdx.y * 64 + wm * 16;
    const int n0   = blockIdx.x * 128 + wn * 64;
    const int lr   = lane & 15;
    const int koffA = (lane < 16) ? 0 : 8;
    const int koffB = (lane < 16) ? 0 : 8;

    const __bf16* pa = A + (size_t)(m0 + lr) * K + koffA;

    v8f acc[4] = {};
    const __bf16* pb[4];
#pragma unroll
    for (int s = 0; s < 4; ++s) {
        int ncl = n0 + s * 16;
        if (ncl > Npad - 16) ncl = Npad - 16;   // clamp: loads always in-bounds
        pb[s] = BT + (size_t)(ncl + lr) * K + koffB;
    }

    for (int kk = 0; kk < K; kk += 32) {
        v16bf a = fragA_ld(pa + kk);
#pragma unroll
        for (int s = 0; s < 4; ++s) {
            v16bf b = fragB_ld(pb[s] + kk);
            acc[s] = __builtin_amdgcn_wmma_f32_16x16x32_bf16(
                false, a, false, b, (short)0, acc[s], false, false);
        }
    }

    const int mb = m0 + ((lane >= 16) ? 8 : 0);
#pragma unroll
    for (int s = 0; s < 4; ++s) {
        int n = n0 + s * 16 + lr;
        if (n0 + s * 16 >= Npad) continue;      // clamped subtile: discard
        if (n >= N) continue;
        float bv = bias ? bias[n] : 0.0f;
#pragma unroll
        for (int e = 0; e < 8; ++e) {
            float v = acc[s][e] + bv;
            if (relu) v = fmaxf(v, 0.0f);
            size_t m = (size_t)(mb + e);
            if (outF) outF[m * (size_t)N + n] = v;
            if (outH) outH[m * (size_t)Npad + n] = (__bf16)v;
        }
    }
}

// ---------------------------------------------------------------------------
// Bidirectional LSTM scan for one layer.
// grid.x = 4: (dir = bid&1, batch-slice of 16 = bid>>1). block = 512 (16 waves).
// Per step: z[16][2048] = h[16][512] @ whT^T (128 WMMA/wave), then gate math.
// h (bf16) and c (f32) live in LDS; z staged through LDS; xg streamed from HBM.
// whT (2 MB/dir) is L2-resident (192 MB L2) so per-step re-reads are cheap.
// Keras gate order i,f,g,o. Backward dir scans t = T2-1..0, writes at t.
// ---------------------------------------------------------------------------
__global__ void k_lstm(const float* __restrict__ xgF, const float* __restrict__ xgB,
                       const __bf16* __restrict__ whTF, const __bf16* __restrict__ whTB,
                       __bf16* __restrict__ hOut) {
    extern __shared__ char smem[];
    float*  zbuf = (float*)smem;                 // 16 x ZLD_ f32
    float*  cbuf = zbuf + 16 * ZLD_;             // 16 x H_  f32
    __bf16* hbuf = (__bf16*)(cbuf + 16 * H_);    // 16 x HLD_ bf16

    const int dir = blockIdx.x & 1;
    const int bs  = blockIdx.x >> 1;
    const float*  xg  = dir ? xgB : xgF;
    const __bf16* whT = dir ? whTB : whTF;

    const int tid  = threadIdx.x;
    const int wid  = tid >> 5;
    const int lane = tid & 31;
    const int lr   = lane & 15;
    const int koffA = (lane < 16) ? 0 : 8;
    const int koffB = (lane < 16) ? 0 : 8;

    for (int i = tid; i < 16 * HLD_; i += 512) hbuf[i] = (__bf16)0.0f;
    for (int i = tid; i < 16 * H_;  i += 512) cbuf[i] = 0.0f;
    __syncthreads();

    const __bf16* pa0 = hbuf + lr * HLD_ + koffA;
    const __bf16* pb0[8];
#pragma unroll
    for (int s = 0; s < 8; ++s)
        pb0[s] = whT + (size_t)(wid * 128 + s * 16 + lr) * H_ + koffB;

    for (int st = 0; st < T2_; ++st) {
        const int t = dir ? (T2_ - 1 - st) : st;

        // z = h @ wh  (each wave: 16 rows x 128 cols, K = 512)
        v8f acc[8] = {};
        for (int ks = 0; ks < H_ / 32; ++ks) {
            v16bf a = fragA_ld(pa0 + ks * 32);
#pragma unroll
            for (int s = 0; s < 8; ++s) {
                v16bf b = fragB_ld(pb0[s] + ks * 32);
                acc[s] = __builtin_amdgcn_wmma_f32_16x16x32_bf16(
                    false, a, false, b, (short)0, acc[s], false, false);
            }
        }

        // write z fragments to LDS
        const int mb = (lane >= 16) ? 8 : 0;
#pragma unroll
        for (int s = 0; s < 8; ++s) {
            int n = wid * 128 + s * 16 + lr;
#pragma unroll
            for (int e = 0; e < 8; ++e)
                zbuf[(mb + e) * ZLD_ + n] = acc[s][e];
        }
        __syncthreads();

        // gate phase: thread j handles hidden column j for all 16 batch rows
        const int j = tid;   // 0..511
        for (int m = 0; m < 16; ++m) {
            size_t rowg = ((size_t)(bs * 16 + m) * T2_ + t) * G4_;
            float zi = zbuf[m * ZLD_ + j]            + xg[rowg + j];
            float zf = zbuf[m * ZLD_ + H_ + j]       + xg[rowg + H_ + j];
            float zg = zbuf[m * ZLD_ + 2 * H_ + j]   + xg[rowg + 2 * H_ + j];
            float zo = zbuf[m * ZLD_ + 3 * H_ + j]   + xg[rowg + 3 * H_ + j];
            float c  = sigmoidf_(zf) * cbuf[m * H_ + j] + sigmoidf_(zi) * tanhf(zg);
            float h  = sigmoidf_(zo) * tanhf(c);
            cbuf[m * H_ + j] = c;
            hbuf[m * HLD_ + j] = (__bf16)h;
            hOut[((size_t)(bs * 16 + m) * T2_ + t) * (2 * H_) + dir * H_ + j] = (__bf16)h;
        }
        __syncthreads();
    }
}

// ---------------------------------------------------------------------------
extern "C" void kernel_launch(void* const* d_in, const int* in_sizes, int n_in,
                              void* d_out, int out_size, void* d_ws, size_t ws_size,
                              hipStream_t stream) {
    (void)in_sizes; (void)n_in; (void)out_size; (void)ws_size;

    const float* x      = (const float*)d_in[0];
    const float* conv_w = (const float*)d_in[1];
    const float* conv_b = (const float*)d_in[2];
    const float* wi1f   = (const float*)d_in[3];
    const float* wh1f   = (const float*)d_in[4];
    const float* b1f    = (const float*)d_in[5];
    const float* wi1b   = (const float*)d_in[6];
    const float* wh1b   = (const float*)d_in[7];
    const float* b1b    = (const float*)d_in[8];
    const float* wi2f   = (const float*)d_in[9];
    const float* wh2f   = (const float*)d_in[10];
    const float* b2f    = (const float*)d_in[11];
    const float* wi2b   = (const float*)d_in[12];
    const float* wh2b   = (const float*)d_in[13];
    const float* b2b    = (const float*)d_in[14];
    const float* dw     = (const float*)d_in[15];
    const float* db     = (const float*)d_in[16];
    float* out = (float*)d_out;

    char* ws = (char*)d_ws;
    size_t off = 0;
    auto alloc = [&](size_t bytes) -> char* {
        char* p = ws + off;
        off = (off + bytes + 255) & ~(size_t)255;
        return p;
    };

    __bf16* convT  = (__bf16*)alloc((size_t)F_ * KCP_ * 2);
    __bf16* wiT1f  = (__bf16*)alloc((size_t)G4_ * F_ * 2);
    __bf16* wiT1b  = (__bf16*)alloc((size_t)G4_ * F_ * 2);
    __bf16* whT1f  = (__bf16*)alloc((size_t)G4_ * H_ * 2);
    __bf16* whT1b  = (__bf16*)alloc((size_t)G4_ * H_ * 2);
    __bf16* wiT2f  = (__bf16*)alloc((size_t)G4_ * 2 * H_ * 2);
    __bf16* wiT2b  = (__bf16*)alloc((size_t)G4_ * 2 * H_ * 2);
    __bf16* whT2f  = (__bf16*)alloc((size_t)G4_ * H_ * 2);
    __bf16* whT2b  = (__bf16*)alloc((size_t)G4_ * H_ * 2);
    __bf16* denseT = (__bf16*)alloc((size_t)32 * 2 * H_ * 2);
    __bf16* col    = (__bf16*)alloc((size_t)MROWS * KCP_ * 2);
    __bf16* convO  = (__bf16*)alloc((size_t)MROWS * F_ * 2);
    float*  xgA    = (float*) alloc((size_t)MROWS * G4_ * 4);
    float*  xgB    = (float*) alloc((size_t)MROWS * G4_ * 4);
    __bf16* h1     = (__bf16*)alloc((size_t)MROWS * 2 * H_ * 2);
    __bf16* h2     = (__bf16*)alloc((size_t)MROWS * 2 * H_ * 2);

    auto launchT = [&](const float* src, __bf16* dst, int K, int N, int Kpad, int Npad) {
        int total = Kpad * Npad;
        k_transpose_bf16<<<dim3((total + 255) / 256), dim3(256), 0, stream>>>(
            src, dst, K, N, Kpad, Npad);
    };

    // Weight prep (f32 -> transposed bf16, zero-padded)
    launchT(conv_w, convT, KC_, F_, KCP_, F_);
    launchT(wi1f, wiT1f, F_, G4_, F_, G4_);
    launchT(wi1b, wiT1b, F_, G4_, F_, G4_);
    launchT(wh1f, whT1f, H_, G4_, H_, G4_);
    launchT(wh1b, whT1b, H_, G4_, H_, G4_);
    launchT(wi2f, wiT2f, 2 * H_, G4_, 2 * H_, G4_);
    launchT(wi2b, wiT2b, 2 * H_, G4_, 2 * H_, G4_);
    launchT(wh2f, whT2f, H_, G4_, H_, G4_);
    launchT(wh2b, whT2b, H_, G4_, H_, G4_);
    launchT(dw, denseT, 2 * H_, V_, 2 * H_, 32);

    // im2col
    {
        int total = MROWS * KCP_;
        k_im2col<<<dim3((total + 255) / 256), dim3(256), 0, stream>>>(x, col);
    }

    // Conv as GEMM: [32000,896] x [896,256] + bias, ReLU -> bf16
    k_gemm<<<dim3(F_ / 128, MROWS / 64), 256, 0, stream>>>(
        col, convT, conv_b, nullptr, convO, F_, F_, KCP_, 1);

    // Layer 1 input projections -> xg f32 [32000][2048]
    k_gemm<<<dim3(G4_ / 128, MROWS / 64), 256, 0, stream>>>(
        convO, wiT1f, b1f, xgA, nullptr, G4_, G4_, F_, 0);
    k_gemm<<<dim3(G4_ / 128, MROWS / 64), 256, 0, stream>>>(
        convO, wiT1b, b1b, xgB, nullptr, G4_, G4_, F_, 0);

    const size_t smemBytes = (size_t)16 * ZLD_ * 4 + (size_t)16 * H_ * 4 + (size_t)16 * HLD_ * 2;

    // Layer 1 bidirectional scan -> h1 bf16 [32000][1024]
    k_lstm<<<dim3(4), dim3(512), smemBytes, stream>>>(xgA, xgB, whT1f, whT1b, h1);

    // Layer 2 input projections (reuse xg buffers)
    k_gemm<<<dim3(G4_ / 128, MROWS / 64), 256, 0, stream>>>(
        h1, wiT2f, b2f, xgA, nullptr, G4_, G4_, 2 * H_, 0);
    k_gemm<<<dim3(G4_ / 128, MROWS / 64), 256, 0, stream>>>(
        h1, wiT2b, b2b, xgB, nullptr, G4_, G4_, 2 * H_, 0);

    // Layer 2 bidirectional scan -> h2
    k_lstm<<<dim3(4), dim3(512), smemBytes, stream>>>(xgA, xgB, whT2f, whT2b, h2);

    // Dense: [32000,1024] x [1024,29] + bias -> f32 out (N padded to 32 in BT)
    k_gemm<<<dim3(1, MROWS / 64), 256, 0, stream>>>(
        h2, denseT, db, out, nullptr, V_, 32, 2 * H_, 0);
}
